// FAVORPlusAttention_13700945674726
// MI455X (gfx1250) — matrix operand order repaired and checked by reading, not compiled, not run on hardware
//
#include <hip/hip_runtime.h>
#include <hip/hip_bf16.h>

// ---------------------------------------------------------------------------
// FAVOR+ attention for MI455X (gfx1250), wave32.
//  - All GEMMs on v_wmma_f32_16x16x32_bf16 (bf16 operands, f32 accumulate)
//  - bf16 tiles staged LDS via GLOBAL_LOAD_ASYNC_TO_LDS_B128 (ASYNCcnt)
//  - fp32->bf16 converting tile fills via VGPR bounce, prefetched ahead
// ---------------------------------------------------------------------------

typedef __attribute__((ext_vector_type(16))) __bf16 v16bf;
typedef __attribute__((ext_vector_type(8)))  float  v8f;

__device__ __forceinline__ v8f vzero8() {
    v8f z = {0.f, 0.f, 0.f, 0.f, 0.f, 0.f, 0.f, 0.f};
    return z;
}

__device__ __forceinline__ unsigned pk_bf16(float a, float b) {
    unsigned short x = __builtin_bit_cast(unsigned short, (__bf16)a);
    unsigned short y = __builtin_bit_cast(unsigned short, (__bf16)b);
    return (unsigned)x | ((unsigned)y << 16);
}

// LDS offset (wave-relative 32-bit) of a generic pointer to __shared__.
__device__ __forceinline__ unsigned lds_off(const void* p) {
    return (unsigned)(unsigned long long)p;
}

// CDNA5 async copy: 16 bytes global -> LDS per lane, tracked by ASYNCcnt.
__device__ __forceinline__ void async_b128(unsigned lds_byte_off, const void* g) {
    asm volatile("global_load_async_to_lds_b128 %0, %1, off"
                 :: "v"(lds_byte_off), "v"((unsigned long long)g)
                 : "memory");
}
__device__ __forceinline__ void wait_async0() {
    asm volatile("s_wait_asynccnt 0x0" ::: "memory");
}

// A fragment: 16x32 bf16 tile, rows M contiguous at stride ld (elements).
// lane m = lane&15, half = lane>>4; dword j -> k0 = (j>>2)*16 + half*8 + (j&3)*2
__device__ __forceinline__ v16bf frag_a(const __bf16* lds, int ld) {
    int lane = threadIdx.x & 31;
    int hl = lane >> 4, m = lane & 15;
    union { v16bf v; unsigned u[8]; } r;
    const __bf16* row = lds + m * ld;
#pragma unroll
    for (int j = 0; j < 8; ++j) {
        int k0 = ((j >> 2) << 4) + hl * 8 + ((j & 3) << 1);
        r.u[j] = *(const unsigned*)(row + k0);
    }
    return r.v;
}

// B fragment: 32x16 bf16 (KxN), LDS holds [n][K] (K contiguous).
// lane n = lane&15, half = lane>>4; dword j -> k0 = (j>>2)*16 + half*8 + (j&3)*2
__device__ __forceinline__ v16bf frag_b(const __bf16* lds, int ld) {
    int lane = threadIdx.x & 31;
    int hl = lane >> 4, n = lane & 15;
    union { v16bf v; unsigned u[8]; } r;
    const __bf16* col = lds + n * ld;
#pragma unroll
    for (int j = 0; j < 8; ++j) {
        int k0 = ((j >> 2) << 4) + hl * 8 + ((j & 3) << 1);
        r.u[j] = *(const unsigned*)(col + k0);
    }
    return r.v;
}

__device__ __forceinline__ v8f wmma_bf16(v16bf a, v16bf b, v8f c) {
    return __builtin_amdgcn_wmma_f32_16x16x32_bf16(
        false, a, false, b, (short)0, c, false, false);
}

// ---------------------------------------------------------------------------
// Generic GEMM: D[M,N] = A[M,K] @ Bw[N,K]^T + bias[N]
// Block tile 128x128, BK=32, 256 threads = 8 waves (2x4), wave = 4x2 tiles.
// A_BF16 path fills the A tile with async-to-LDS B128 copies.
// ---------------------------------------------------------------------------
template <bool A_BF16, bool OUT_BF16>
__global__ __launch_bounds__(256) void gemm_bias_kernel(
    const void* __restrict__ Ap, const float* __restrict__ Bw,
    const float* __restrict__ bias, void* __restrict__ Dp,
    int M, int Nout, int K)
{
    constexpr int LDA = 40, LDB = 40;   // 80-byte row stride: 16B aligned, 16 banks
    __shared__ __align__(16) __bf16 sA[128 * LDA];
    __shared__ __align__(16) __bf16 sB[128 * LDB];

    const int rowBase = blockIdx.x * 128;
    const int colBase = blockIdx.y * 128;
    const int tid = threadIdx.x;
    const int wid = tid >> 5, lane = tid & 31;
    const int wm = wid >> 2, wn = wid & 3;
    const int lrow = tid >> 1, lseg = tid & 1;

    v8f acc[4][2];
#pragma unroll
    for (int mi = 0; mi < 4; ++mi)
#pragma unroll
        for (int ni = 0; ni < 2; ++ni) acc[mi][ni] = vzero8();

    for (int kk = 0; kk < K; kk += 32) {
        // --- A tile: rows rowBase..+127, K slice kk..kk+31 ---
        if (A_BF16) {
            const __bf16* g = (const __bf16*)Ap
                            + (size_t)(rowBase + lrow) * K + kk + lseg * 16;
            unsigned d0 = lds_off(sA + lrow * LDA + lseg * 16);
            async_b128(d0,      g);          // 8 bf16
            async_b128(d0 + 16, g + 8);      // 8 bf16
        } else {
            const float* Af = (const float*)Ap;
            const float* gp = Af + (size_t)(rowBase + lrow) * K + kk + lseg * 16;
            if (kk + 32 < K) __builtin_prefetch(gp + 32);   // global_prefetch_b8
            const float4* src = (const float4*)gp;
            unsigned* dst = (unsigned*)(sA + lrow * LDA + lseg * 16);
#pragma unroll
            for (int j = 0; j < 4; ++j) {
                float4 f = src[j];
                dst[2 * j]     = pk_bf16(f.x, f.y);
                dst[2 * j + 1] = pk_bf16(f.z, f.w);
            }
        }
        // --- B tile: weight rows colBase..+127 (fp32, K contiguous) ---
        {
            const float* gp = Bw + (size_t)(colBase + lrow) * K + kk + lseg * 16;
            if (kk + 32 < K) __builtin_prefetch(gp + 32);
            const float4* src = (const float4*)gp;
            unsigned* dst = (unsigned*)(sB + lrow * LDB + lseg * 16);
#pragma unroll
            for (int j = 0; j < 4; ++j) {
                float4 f = src[j];
                dst[2 * j]     = pk_bf16(f.x, f.y);
                dst[2 * j + 1] = pk_bf16(f.z, f.w);
            }
        }
        if (A_BF16) wait_async0();
        __syncthreads();

        v16bf af[4], bf2[2];
#pragma unroll
        for (int mi = 0; mi < 4; ++mi)
            af[mi] = frag_a(sA + (wm * 64 + mi * 16) * LDA, LDA);
#pragma unroll
        for (int ni = 0; ni < 2; ++ni)
            bf2[ni] = frag_b(sB + (wn * 32 + ni * 16) * LDB, LDB);
#pragma unroll
        for (int mi = 0; mi < 4; ++mi)
#pragma unroll
            for (int ni = 0; ni < 2; ++ni)
                acc[mi][ni] = wmma_bf16(af[mi], bf2[ni], acc[mi][ni]);
        __syncthreads();
    }

    const int hl = lane >> 4, lcol = lane & 15;
#pragma unroll
    for (int mi = 0; mi < 4; ++mi) {
#pragma unroll
        for (int ni = 0; ni < 2; ++ni) {
            const int col = colBase + wn * 32 + ni * 16 + lcol;
            const float bv = bias[col];
#pragma unroll
            for (int i = 0; i < 8; ++i) {
                const int row = rowBase + wm * 64 + mi * 16 + hl * 8 + i;
                const float v = acc[mi][ni][i] + bv;
                if (OUT_BF16)
                    ((__bf16*)Dp)[(size_t)row * Nout + col] = (__bf16)v;
                else
                    ((float*)Dp)[(size_t)row * Nout + col] = v;
            }
        }
    }
}

// ---------------------------------------------------------------------------
// Phi kernel: per (b,h): proj = q_or_k[128 x 64] @ RM[256 x 64]^T;
// phi = exp(proj - 0.5*|q|^2)/16.  Grid (64 heads, 32 n-tiles).
// q/k tile (bf16) arrives via async-to-LDS; RM converted fp32->bf16 once.
// ---------------------------------------------------------------------------
__global__ __launch_bounds__(256) void phi_kernel(
    const __bf16* __restrict__ qkv, const float* __restrict__ RM,
    __bf16* __restrict__ phi, int sel)
{
    constexpr int LDA = 72, LDB = 68;   // A: 144B rows (16B aligned)
    __shared__ __align__(16) __bf16 sA[128 * LDA];   // q/k rows [n][d]
    __shared__ __align__(16) __bf16 sB[256 * LDB];   // RM rows  [r][d]
    __shared__ float sNorm[128];

    const int bh = blockIdx.x;
    const int b = bh >> 4, h = bh & 15;
    const int nBase = blockIdx.y * 128;
    const int tid = threadIdx.x;
    const int wid = tid >> 5, lane = tid & 31;
    const int wm = wid >> 2, wn = wid & 3;

    {   // async A: 2 threads/row, 32 bf16 (64B) each = 4 B128 chunks
        const int row = tid >> 1, seg = tid & 1;
        const __bf16* g = qkv + ((size_t)(b * 4096 + nBase + row)) * 3072
                        + (size_t)sel * 1024 + h * 64 + seg * 32;
        unsigned d0 = lds_off(sA + row * LDA + seg * 32);
#pragma unroll
        for (int j = 0; j < 4; ++j)
            async_b128(d0 + j * 16, g + j * 8);
    }
    {   // B: random matrix 256x64 fp32 -> bf16, 1 thread per row
        const float4* src = (const float4*)(RM + tid * 64);
        unsigned* dst = (unsigned*)(sB + tid * LDB);
#pragma unroll
        for (int j = 0; j < 16; ++j) {
            float4 f = src[j];
            dst[2 * j]     = pk_bf16(f.x, f.y);
            dst[2 * j + 1] = pk_bf16(f.z, f.w);
        }
    }
    wait_async0();
    __syncthreads();
    if (tid < 128) {
        const __bf16* rowp = sA + tid * LDA;
        float s = 0.f;
#pragma unroll 8
        for (int d = 0; d < 64; ++d) { float q = (float)rowp[d]; s += q * q; }
        sNorm[tid] = 0.5f * s;
    }
    __syncthreads();

    const int hl = lane >> 4, lcol = lane & 15;
    for (int nh = 0; nh < 2; ++nh) {
        v8f acc[4][2];
#pragma unroll
        for (int mi = 0; mi < 4; ++mi)
#pragma unroll
            for (int ni = 0; ni < 2; ++ni) acc[mi][ni] = vzero8();

#pragma unroll
        for (int ks = 0; ks < 2; ++ks) {
            v16bf af[4], bf2[2];
#pragma unroll
            for (int mi = 0; mi < 4; ++mi)
                af[mi] = frag_a(sA + (wm * 64 + mi * 16) * LDA + ks * 32, LDA);
#pragma unroll
            for (int ni = 0; ni < 2; ++ni)
                bf2[ni] = frag_b(sB + (nh * 128 + wn * 32 + ni * 16) * LDB + ks * 32, LDB);
#pragma unroll
            for (int mi = 0; mi < 4; ++mi)
#pragma unroll
                for (int ni = 0; ni < 2; ++ni)
                    acc[mi][ni] = wmma_bf16(af[mi], bf2[ni], acc[mi][ni]);
        }
#pragma unroll
        for (int mi = 0; mi < 4; ++mi) {
#pragma unroll
            for (int ni = 0; ni < 2; ++ni) {
                const int col = nh * 128 + wn * 32 + ni * 16 + lcol;
#pragma unroll
                for (int i = 0; i < 8; ++i) {
                    const int row = wm * 64 + mi * 16 + hl * 8 + i;
                    const float val =
                        __expf(acc[mi][ni][i] - sNorm[row]) * 0.0625f;
                    phi[((size_t)bh * 4096 + nBase + row) * 256 + col] = (__bf16)val;
                }
            }
        }
    }
}

// ---------------------------------------------------------------------------
// kv kernel: per (b,h): kv[256 r x 64 d] = phi_k^T @ v over N=4096;
// denom[r] = sum_n phi_k.  kv stored transposed as kvT[d][r] (fp32).
// Transposed tile fills stay on the VGPR path (async mover can't transpose).
// ---------------------------------------------------------------------------
__global__ __launch_bounds__(256) void kv_kernel(
    const __bf16* __restrict__ phi_k, const __bf16* __restrict__ qkv,
    float* __restrict__ kvT, float* __restrict__ denom)
{
    constexpr int LDA = 34, LDB = 34;
    __shared__ __align__(16) __bf16 sA[256 * LDA];   // [r][n] (transposed)
    __shared__ __align__(16) __bf16 sB[64 * LDB];    // [d][n] (transposed)

    const int bh = blockIdx.x;
    const int b = bh >> 4, h = bh & 15;
    const int tid = threadIdx.x;
    const int wid = tid >> 5, lane = tid & 31;

    float denAcc = 0.f;
    v8f acc[2][4];
#pragma unroll
    for (int mi = 0; mi < 2; ++mi)
#pragma unroll
        for (int ni = 0; ni < 4; ++ni) acc[mi][ni] = vzero8();

    for (int kk = 0; kk < 4096; kk += 32) {
        {   // phi_k tile [32 n x 256 r] -> transposed LDS
            const int n = tid & 31, rb = (tid >> 5) * 32;
            const __bf16* src = phi_k + ((size_t)bh * 4096 + kk + n) * 256 + rb;
            if (kk + 32 < 4096) __builtin_prefetch(src + 32 * 256);
#pragma unroll
            for (int j = 0; j < 32; ++j) sA[(rb + j) * LDA + n] = src[j];
        }
        {   // v tile [32 n x 64 d] -> transposed LDS
            const int n = tid & 31, db = (tid >> 5) * 8;
            const __bf16* src =
                qkv + ((size_t)(b * 4096 + kk + n)) * 3072 + 2048 + h * 64 + db;
            if (kk + 32 < 4096) __builtin_prefetch(src + (size_t)32 * 3072);
#pragma unroll
            for (int j = 0; j < 8; ++j) sB[(db + j) * LDB + n] = src[j];
        }
        __syncthreads();
        {   // denom partial: thread tid owns r = tid
            const __bf16* rowp = sA + tid * LDA;
            float s = 0.f;
#pragma unroll 8
            for (int j = 0; j < 32; ++j) s += (float)rowp[j];
            denAcc += s;
        }
        v16bf a0 = frag_a(sA + (wid * 32 + 0) * LDA, LDA);
        v16bf a1 = frag_a(sA + (wid * 32 + 16) * LDA, LDA);
        v16bf bb[4];
#pragma unroll
        for (int ni = 0; ni < 4; ++ni) bb[ni] = frag_b(sB + ni * 16 * LDB, LDB);
#pragma unroll
        for (int ni = 0; ni < 4; ++ni) {
            acc[0][ni] = wmma_bf16(a0, bb[ni], acc[0][ni]);
            acc[1][ni] = wmma_bf16(a1, bb[ni], acc[1][ni]);
        }
        __syncthreads();
    }

    denom[bh * 256 + tid] = denAcc;

    const int hl = lane >> 4, lcol = lane & 15;
#pragma unroll
    for (int mi = 0; mi < 2; ++mi)
#pragma unroll
        for (int ni = 0; ni < 4; ++ni)
#pragma unroll
            for (int i = 0; i < 8; ++i) {
                const int r = wid * 32 + mi * 16 + hl * 8 + i;
                const int d = ni * 16 + lcol;
                kvT[(size_t)bh * 16384 + d * 256 + r] = acc[mi][ni][i];
            }
}

// ---------------------------------------------------------------------------
// Output-numerator kernel: per (b,h): num[128 n x 64 d] = phi_q @ kv,
// den[n] = phi_q . denom, out = num/(den+1e-6) -> attn[B,N,H*D] bf16.
// phi_q tile (bf16, K-contiguous) arrives via async-to-LDS.
// ---------------------------------------------------------------------------
__global__ __launch_bounds__(256) void favor_out_kernel(
    const __bf16* __restrict__ phi_q, const float* __restrict__ kvT,
    const float* __restrict__ denom, __bf16* __restrict__ attn)
{
    constexpr int LDA = 40, LDB = 40;
    __shared__ __align__(16) __bf16 sA[128 * LDA];   // phi_q [n][r-chunk]
    __shared__ __align__(16) __bf16 sB[64 * LDB];    // kvT   [d][r-chunk]
    __shared__ float sDen[32];
    __shared__ float sDenRow[128];

    const int bh = blockIdx.x;
    const int b = bh >> 4, h = bh & 15;
    const int nBase = blockIdx.y * 128;
    const int tid = threadIdx.x;
    const int wid = tid >> 5, lane = tid & 31;

    float denAcc = 0.f;
    v8f acc[4];
#pragma unroll
    for (int ni = 0; ni < 4; ++ni) acc[ni] = vzero8();

    for (int kk = 0; kk < 256; kk += 32) {
        {   // async phi_q tile: 2 threads/row, 16 bf16 (32B) each
            const int row = tid >> 1, seg = tid & 1;
            const __bf16* g =
                phi_q + ((size_t)bh * 4096 + nBase + row) * 256 + kk + seg * 16;
            unsigned d0 = lds_off(sA + row * LDA + seg * 16);
            async_b128(d0,      g);
            async_b128(d0 + 16, g + 8);
        }
        {   // kvT tile fp32 [d][r] -> bf16 (K contiguous)
            const int d = tid >> 2, seg = tid & 3;
            const float* src = kvT + (size_t)bh * 16384 + d * 256 + kk + seg * 8;
            unsigned* dst = (unsigned*)(sB + d * LDB + seg * 8);
#pragma unroll
            for (int j = 0; j < 4; ++j) dst[j] = pk_bf16(src[2 * j], src[2 * j + 1]);
        }
        if (tid < 32) sDen[tid] = denom[bh * 256 + kk + tid];
        wait_async0();
        __syncthreads();

        if (tid < 128) {   // den dot-product from resident A tile
            const __bf16* rowp = sA + tid * LDA;
            float s = 0.f;
#pragma unroll 8
            for (int j = 0; j < 32; ++j) s += (float)rowp[j] * sDen[j];
            denAcc += s;
        }
        v16bf a = frag_a(sA + wid * 16 * LDA, LDA);
#pragma unroll
        for (int ni = 0; ni < 4; ++ni) {
            v16bf bb = frag_b(sB + ni * 16 * LDB, LDB);
            acc[ni] = wmma_bf16(a, bb, acc[ni]);
        }
        __syncthreads();
    }

    if (tid < 128) sDenRow[tid] = denAcc;
    __syncthreads();

    const int hl = lane >> 4, lcol = lane & 15;
#pragma unroll
    for (int ni = 0; ni < 4; ++ni)
#pragma unroll
        for (int i = 0; i < 8; ++i) {
            const int row = wid * 16 + hl * 8 + i;
            const int d = ni * 16 + lcol;
            const float v = acc[ni][i] / (sDenRow[row] + 1e-6f);
            attn[((size_t)(b * 4096) + nBase + row) * 1024 + h * 64 + d] = (__bf16)v;
        }
}

// ---------------------------------------------------------------------------
// Launch: x, Wqkv, bqkv, Wproj, bproj, random_matrix  (setup_inputs order)
// Workspace layout (bytes):
//   qkv   bf16 [16384,3072]  @ 0          (100,663,296)
//   phi_q bf16 [64,4096,256] @ 100663296  (134,217,728)
//   phi_k bf16 [64,4096,256] @ 234881024  (134,217,728)
//   kvT   f32  [64,64,256]   @ 369098752  (  4,194,304)
//   denom f32  [64,256]      @ 373293056  (     65,536)
//   attn  bf16 [16384,1024]  @ 373358592  ( 33,554,432)   total ~388 MB
// ---------------------------------------------------------------------------
extern "C" void kernel_launch(void* const* d_in, const int* in_sizes, int n_in,
                              void* d_out, int out_size, void* d_ws, size_t ws_size,
                              hipStream_t stream)
{
    const float* x     = (const float*)d_in[0];
    const float* Wqkv  = (const float*)d_in[1];
    const float* bqkv  = (const float*)d_in[2];
    const float* Wproj = (const float*)d_in[3];
    const float* bproj = (const float*)d_in[4];
    const float* RM    = (const float*)d_in[5];
    float* out = (float*)d_out;

    char* ws = (char*)d_ws;
    __bf16* qkv   = (__bf16*)(ws);
    __bf16* phi_q = (__bf16*)(ws + 100663296ull);
    __bf16* phi_k = (__bf16*)(ws + 234881024ull);
    float*  kvT   = (float*) (ws + 369098752ull);
    float*  den   = (float*) (ws + 373293056ull);
    __bf16* attn  = (__bf16*)(ws + 373358592ull);

    const dim3 blk(256);

    // 1) qkv = x @ Wqkv^T + bqkv         (M=16384, N=3072, K=1024)
    gemm_bias_kernel<false, true><<<dim3(128, 24), blk, 0, stream>>>(
        x, Wqkv, bqkv, qkv, 16384, 3072, 1024);

    // 2) phi_q, phi_k
    phi_kernel<<<dim3(64, 32), blk, 0, stream>>>(qkv, RM, phi_q, 0);
    phi_kernel<<<dim3(64, 32), blk, 0, stream>>>(qkv, RM, phi_k, 1);

    // 3) kvT, denom
    kv_kernel<<<dim3(64), blk, 0, stream>>>(phi_k, qkv, kvT, den);

    // 4) attn = (phi_q @ kv) / (phi_q . denom + 1e-6)
    favor_out_kernel<<<dim3(64, 32), blk, 0, stream>>>(phi_q, kvT, den, attn);

    // 5) out = attn @ Wproj^T + bproj    (M=16384, N=1024, K=1024)
    gemm_bias_kernel<true, false><<<dim3(128, 8), blk, 0, stream>>>(
        attn, Wproj, bproj, out, 16384, 1024, 1024);
}
